// SlidingWindowAttention_44925357916184
// MI455X (gfx1250) — matrix operand order repaired and checked by reading, not compiled, not run on hardware
//
#include <hip/hip_runtime.h>
#include <cstdint>

// ---------------- types ----------------
typedef __attribute__((ext_vector_type(16))) __bf16 v16bf;
typedef __attribute__((ext_vector_type(2)))  __bf16 bf16x2;
typedef __attribute__((ext_vector_type(8)))  float  v8f;

// ---------------- problem constants ----------------
#define BB     2
#define SS     2048
#define DD     2048
#define HD     128
#define NH     16
#define NKV    4
#define KVD    512          // NKV*HD
#define NQKV   3072         // DD + 2*KVD
#define MM     4096         // BB*SS
#define WIN    512

// ======================================================================
// fp32 -> bf16 weight packing: Wqkv[k][n], n<2048:wq, <2560:wk, else wv
// ======================================================================
__global__ __launch_bounds__(256) void swa_pack_wqkv(const float* __restrict__ wq,
                                                     const float* __restrict__ wk,
                                                     const float* __restrict__ wv,
                                                     __bf16* __restrict__ Wqkv) {
    size_t idx = (size_t)blockIdx.x * 256 + threadIdx.x;
    size_t total = (size_t)DD * NQKV;
    if (idx >= total) return;
    int    n = (int)(idx % NQKV);
    size_t k = idx / NQKV;
    float v;
    if (n < DD)            v = wq[k * DD + n];
    else if (n < DD + KVD) v = wk[k * KVD + (n - DD)];
    else                   v = wv[k * KVD + (n - DD - KVD)];
    Wqkv[idx] = (__bf16)v;
}

__global__ __launch_bounds__(256) void swa_pack_wo(const float* __restrict__ wo,
                                                   __bf16* __restrict__ Wo) {
    size_t idx = (size_t)blockIdx.x * 256 + threadIdx.x;
    if (idx >= (size_t)DD * DD) return;
    Wo[idx] = (__bf16)wo[idx];
}

// ======================================================================
// RMSNorm: one 256-thread block per token row, write bf16
// ======================================================================
__global__ __launch_bounds__(256) void swa_rmsnorm(const float* __restrict__ x,
                                                   const float* __restrict__ w,
                                                   __bf16* __restrict__ xn) {
    __shared__ float red[256];
    const int tid = threadIdx.x;
    const size_t row = blockIdx.x;
    const float* xr = x + row * DD;
    float s = 0.0f;
    for (int i = tid; i < DD; i += 256) { float v = xr[i]; s += v * v; }
    red[tid] = s;
    __syncthreads();
    for (int st = 128; st > 0; st >>= 1) {
        if (tid < st) red[tid] += red[tid + st];
        __syncthreads();
    }
    const float scale = rsqrtf(red[0] * (1.0f / DD) + 1e-6f);
    for (int i = tid; i < DD; i += 256)
        xn[row * DD + i] = (__bf16)(xr[i] * scale * w[i]);
}

// ======================================================================
// WMMA bf16 GEMM: C[M,N] = A[M,K] x B[K,N], row-major.
// One wave computes a 32x64 output strip: 2 (M) x 4 (N) 16x16 subtiles.
// Per K-step of 32: 2 A operands + 4 B operands feed 8
// v_wmma_f32_16x16x32_bf16 -> each A reused 4x, each B reused 2x.
// Output to bf16 (Cb) or fp32 (Cf).
// ======================================================================
__global__ __launch_bounds__(256) void swa_wmma_gemm(const __bf16* __restrict__ A,
                                                     const __bf16* __restrict__ Bm,
                                                     __bf16* Cb, float* Cf,
                                                     int Md, int Nd, int Kd) {
    const int wave = threadIdx.x >> 5;
    const int lane = threadIdx.x & 31;
    const int tcols = Nd >> 6;                       // N / 64
    const int tile = blockIdx.x * 8 + wave;
    if (tile >= (Md >> 5) * tcols) return;           // uniform per wave -> EXEC all-ones
    const int m0 = (tile / tcols) << 5;
    const int n0 = (tile % tcols) << 6;

    // A layout (16x32 bf16): lane L<16 row M=L K-halves {0..7,16..23};
    // lanes 16..31 same rows, K offset +8.
    const int a_row  = lane & 15;
    const int a_koff = (lane >> 4) << 3;
    const __bf16* Arow0 = A + (size_t)(m0 + a_row) * Kd;
    const __bf16* Arow1 = A + (size_t)(m0 + 16 + a_row) * Kd;

    v8f acc[2][4];
#pragma unroll
    for (int mi = 0; mi < 2; ++mi)
#pragma unroll
        for (int ni = 0; ni < 4; ++ni) acc[mi][ni] = (v8f){};

    for (int kk = 0; kk < Kd; kk += 32) {
        __builtin_prefetch(Arow0 + kk + 256, 0, 1);  // global_prefetch_b8
        __builtin_prefetch(Arow1 + kk + 256, 0, 1);
        v16bf av0, av1;
#pragma unroll
        for (int j = 0; j < 8; ++j) {
            const int k2 = (j < 4 ? 2 * j : 16 + 2 * (j - 4)) + a_koff;
            bf16x2 p0 = *(const bf16x2*)(Arow0 + kk + k2);
            bf16x2 p1 = *(const bf16x2*)(Arow1 + kk + k2);
            av0[2 * j] = p0[0]; av0[2 * j + 1] = p0[1];
            av1[2 * j] = p1[0]; av1[2 * j + 1] = p1[1];
        }
        // B layout (32x16): lane&15 = column N, elements = K in same order as A
        const __bf16* Brow = Bm + (size_t)kk * Nd + n0 + (lane & 15);
#pragma unroll
        for (int ni = 0; ni < 4; ++ni) {
            v16bf bv;
#pragma unroll
            for (int j = 0; j < 8; ++j) {
                const int k2 = (j < 4 ? 2 * j : 16 + 2 * (j - 4)) + a_koff;
                bv[2 * j] = Brow[(size_t)k2 * Nd + ni * 16]; bv[2 * j + 1] = Brow[(size_t)(k2 + 1) * Nd + ni * 16];
            }
            acc[0][ni] = __builtin_amdgcn_wmma_f32_16x16x32_bf16(false, av0, false, bv,
                                                                 (short)0, acc[0][ni],
                                                                 false, false);
            acc[1][ni] = __builtin_amdgcn_wmma_f32_16x16x32_bf16(false, av1, false, bv,
                                                                 (short)0, acc[1][ni],
                                                                 false, false);
        }
    }
    // C layout: vgpr r, lanes 0-15 -> M=r, lanes 16-31 -> M=r+8; N=lane&15
    const int col = lane & 15;
    const int rowoff = (lane >> 4) << 3;
#pragma unroll
    for (int mi = 0; mi < 2; ++mi) {
#pragma unroll
        for (int ni = 0; ni < 4; ++ni) {
#pragma unroll
            for (int r = 0; r < 8; ++r) {
                const size_t o = (size_t)(m0 + mi * 16 + r + rowoff) * Nd
                               + n0 + ni * 16 + col;
                if (Cf) Cf[o] = acc[mi][ni][r];
                else    Cb[o] = (__bf16)acc[mi][ni][r];
            }
        }
    }
}

// ======================================================================
// RoPE + layout transform
// ======================================================================
__global__ __launch_bounds__(256) void swa_rope_q(const __bf16* __restrict__ QKV,
                                                  __bf16* __restrict__ Qr) {
    const int idx = blockIdx.x * 256 + threadIdx.x;       // B*S*NH*64 = 2^22
    const int i = idx & 63;
    const int h = (idx >> 6) & 15;
    const int s = (idx >> 10) & 2047;
    const int b = idx >> 21;
    const __bf16* src = QKV + ((size_t)(b * SS + s)) * NQKV + h * HD + 2 * i;
    const float x1 = (float)src[0], x2 = (float)src[1];
    const float ang = (float)s * __powf(10000.0f, -(float)(2 * i) * (1.0f / 128.0f));
    const float c = __cosf(ang), sn = __sinf(ang);
    __bf16* dst = Qr + (((size_t)(b * NH + h)) * SS + s) * HD + 2 * i;
    dst[0] = (__bf16)(x1 * c - x2 * sn);
    dst[1] = (__bf16)(x1 * sn + x2 * c);
}

__global__ __launch_bounds__(256) void swa_rope_kv(const __bf16* __restrict__ QKV,
                                                   __bf16* __restrict__ Kr,
                                                   __bf16* __restrict__ Vc) {
    const int idx = blockIdx.x * 256 + threadIdx.x;       // B*S*NKV*64 = 2^20
    const int i = idx & 63;
    const int h = (idx >> 6) & 3;
    const int s = (idx >> 8) & 2047;
    const int b = idx >> 19;
    const size_t row = (size_t)(b * SS + s) * NQKV;
    const __bf16* srcK = QKV + row + DD + h * HD + 2 * i;
    const __bf16* srcV = QKV + row + DD + KVD + h * HD + 2 * i;
    const float x1 = (float)srcK[0], x2 = (float)srcK[1];
    const float ang = (float)s * __powf(10000.0f, -(float)(2 * i) * (1.0f / 128.0f));
    const float c = __cosf(ang), sn = __sinf(ang);
    const size_t dofs = (((size_t)(b * NKV + h)) * SS + s) * HD + 2 * i;
    Kr[dofs]     = (__bf16)(x1 * c - x2 * sn);
    Kr[dofs + 1] = (__bf16)(x1 * sn + x2 * c);
    Vc[dofs]     = srcV[0];
    Vc[dofs + 1] = srcV[1];
}

// ======================================================================
// Sliding-window flash attention: one wave (32 threads) per
// (batch, head, 16-query tile). K/V tiles staged in LDS; QK^T and PV on
// v_wmma_f32_16x16x32_bf16; online softmax via 16-lane shfl_xor.
// ======================================================================
__global__ __launch_bounds__(32) void swa_attn(const __bf16* __restrict__ Qr,
                                               const __bf16* __restrict__ Kr,
                                               const __bf16* __restrict__ Vc,
                                               __bf16* __restrict__ attn) {
    __shared__ __align__(16) __bf16 lK[16 * 128];
    __shared__ __align__(16) __bf16 lV[16 * 128];
    __shared__ __align__(16) __bf16 lP[16 * 32];

    const int lane = threadIdx.x;
    const int qt = blockIdx.x & 127;
    const int h  = (blockIdx.x >> 7) & 15;
    const int b  = blockIdx.x >> 11;
    const int kvh = h >> 2;                  // n_rep = 4
    const int q0 = qt << 4;

    const __bf16* Qbase = Qr + (((size_t)(b * NH + h)) * SS + q0) * HD;
    const __bf16* Kbase = Kr + ((size_t)(b * NKV + kvh)) * SS * HD;
    const __bf16* Vbase = Vc + ((size_t)(b * NKV + kvh)) * SS * HD;

    const int a_row  = lane & 15;
    const int a_koff = (lane >> 4) << 3;

    // Q tile 16x128 as four 16x32 A operands, resident for whole loop
    v16bf aQ[4];
#pragma unroll
    for (int c = 0; c < 4; ++c) {
#pragma unroll
        for (int j = 0; j < 8; ++j) {
            const int k2 = (j < 4 ? 2 * j : 16 + 2 * (j - 4)) + a_koff;
            bf16x2 p = *(const bf16x2*)(Qbase + (size_t)a_row * HD + c * 32 + k2);
            aQ[c][2 * j] = p[0]; aQ[c][2 * j + 1] = p[1];
        }
    }

    float m_i[8], l_i[8];
    v8f O[8];
#pragma unroll
    for (int r = 0; r < 8; ++r) { m_i[r] = -3.0e38f; l_i[r] = 0.0f; }
#pragma unroll
    for (int n = 0; n < 8; ++n) O[n] = (v8f){};

    const int col = lane & 15;
    const int rowoff = (lane >> 4) << 3;
    const float scale = 0.088388347648318447f;      // 1/sqrt(128)

    int kstart = q0 - WIN; if (kstart < 0) kstart = 0;
    for (int k0 = kstart; k0 <= q0; k0 += 16) {
        // ---- stage 16x128 K and V tiles into LDS (b128 copies) ----
        const int4* Ks = (const int4*)(Kbase + (size_t)k0 * HD);
        const int4* Vs = (const int4*)(Vbase + (size_t)k0 * HD);
        int4* dK = (int4*)lK; int4* dV = (int4*)lV;
#pragma unroll
        for (int it = 0; it < 8; ++it) {
            dK[it * 32 + lane] = Ks[it * 32 + lane];
            dV[it * 32 + lane] = Vs[it * 32 + lane];
        }
        __syncthreads();

        // ---- S = Q . K^T (inner dim 128 = 4 chained WMMAs) ----
        v8f sacc = {};
#pragma unroll
        for (int c = 0; c < 4; ++c) {
            v16bf bv;       // B operand: lane&15 = key, elements = hd-index in same order as A
#pragma unroll
            for (int j = 0; j < 8; ++j) {
                const int k2 = (j < 4 ? 2 * j : 16 + 2 * (j - 4)) + a_koff;
                bv[2 * j]     = lK[(lane & 15) * 128 + c * 32 + k2];
                bv[2 * j + 1] = lK[(lane & 15) * 128 + c * 32 + k2 + 1];
            }
            sacc = __builtin_amdgcn_wmma_f32_16x16x32_bf16(false, aQ[c], false, bv,
                                                           (short)0, sacc, false, false);
        }

        // ---- mask + scale (C layout: row = r+rowoff, key = k0+col) ----
#pragma unroll
        for (int r = 0; r < 8; ++r) {
            const int q = q0 + r + rowoff;
            const int k = k0 + col;
            const float sv = sacc[r] * scale;
            const bool ok = (k <= q) && (q - k <= WIN);
            sacc[r] = ok ? sv : -3.0e38f;
        }

        // ---- online softmax: rows live in 16-lane half-wave groups ----
#pragma unroll
        for (int r = 0; r < 8; ++r) {
            float mb = sacc[r];
#pragma unroll
            for (int off = 1; off < 16; off <<= 1) mb = fmaxf(mb, __shfl_xor(mb, off, 32));
            const float mnew = fmaxf(m_i[r], mb);
            const float p = __expf(sacc[r] - mnew);
            sacc[r] = p;
            float lb = p;
#pragma unroll
            for (int off = 1; off < 16; off <<= 1) lb += __shfl_xor(lb, off, 32);
            const float corr = __expf(m_i[r] - mnew);
            l_i[r] = l_i[r] * corr + lb;
            m_i[r] = mnew;
#pragma unroll
            for (int n = 0; n < 8; ++n) O[n][r] *= corr;
        }

        // ---- bounce P through LDS to convert C-layout -> A-layout (zero-pad to K=32) ----
#pragma unroll
        for (int r = 0; r < 8; ++r) {
            lP[(r + rowoff) * 32 + col]      = (__bf16)sacc[r];
            lP[(r + rowoff) * 32 + 16 + col] = (__bf16)0.0f;
        }
        __syncthreads();

        v16bf aP;
#pragma unroll
        for (int j = 0; j < 8; ++j) {
            const int k2 = (j < 4 ? 2 * j : 16 + 2 * (j - 4)) + a_koff;
            bf16x2 p = *(const bf16x2*)(lP + a_row * 32 + k2);
            aP[2 * j] = p[0]; aP[2 * j + 1] = p[1];
        }

        // ---- O += P . V  (K padded to 32: elements 8..15 (k 16..31) carry zeros) ----
        const int krow = lane & 15;
        const bool kvalid = lane < 16;
#pragma unroll
        for (int n = 0; n < 8; ++n) {
            v16bf bv;
#pragma unroll
            for (int j = 0; j < 8; ++j) {
                const int k2 = (j < 4 ? 2 * j : 16 + 2 * (j - 4)) + a_koff;
                bv[2 * j]     = (j < 4) ? lV[k2 * 128 + n * 16 + col] : (__bf16)0.0f;
                bv[2 * j + 1] = (j < 4) ? lV[(k2 + 1) * 128 + n * 16 + col] : (__bf16)0.0f;
            }
            O[n] = __builtin_amdgcn_wmma_f32_16x16x32_bf16(false, aP, false, bv,
                                                           (short)0, O[n], false, false);
        }
        __syncthreads();
    }

    // ---- normalize and store to [B,S,D] bf16 ----
#pragma unroll
    for (int r = 0; r < 8; ++r) {
        const float inv = 1.0f / l_i[r];
        const int q = q0 + r + rowoff;
        const size_t base = ((size_t)(b * SS + q)) * DD + h * HD;
#pragma unroll
        for (int n = 0; n < 8; ++n)
            attn[base + n * 16 + col] = (__bf16)(O[n][r] * inv);
    }
}

// ======================================================================
// launch
// ======================================================================
extern "C" void kernel_launch(void* const* d_in, const int* in_sizes, int n_in,
                              void* d_out, int out_size, void* d_ws, size_t ws_size,
                              hipStream_t stream) {
    (void)in_sizes; (void)n_in; (void)out_size; (void)ws_size;
    const float* x     = (const float*)d_in[0];
    const float* rms_w = (const float*)d_in[1];
    const float* wq    = (const float*)d_in[2];
    const float* wk    = (const float*)d_in[3];
    const float* wv    = (const float*)d_in[4];
    const float* wo    = (const float*)d_in[5];
    float* out = (float*)d_out;

    // workspace layout (all bf16), ~105 MB total — fits comfortably in L2
    __bf16* Wqkv = (__bf16*)d_ws;
    __bf16* Wo   = Wqkv + (size_t)DD * NQKV;
    __bf16* Xn   = Wo   + (size_t)DD * DD;
    __bf16* QKV  = Xn   + (size_t)MM * DD;
    __bf16* Qr   = QKV  + (size_t)MM * NQKV;
    __bf16* Krp  = Qr   + (size_t)BB * NH  * SS * HD;
    __bf16* Vcp  = Krp  + (size_t)BB * NKV * SS * HD;
    __bf16* Attn = Vcp  + (size_t)BB * NKV * SS * HD;

    swa_pack_wqkv<<<(DD * NQKV) / 256, 256, 0, stream>>>(wq, wk, wv, Wqkv);
    swa_pack_wo<<<(DD * DD) / 256, 256, 0, stream>>>(wo, Wo);
    swa_rmsnorm<<<MM, 256, 0, stream>>>(x, rms_w, Xn);

    // QKV = Xn @ Wqkv : macro-tiles = (4096/32)*(3072/64) = 6144, 8 waves/block
    swa_wmma_gemm<<<6144 / 8, 256, 0, stream>>>(Xn, Wqkv, QKV, nullptr,
                                                MM, NQKV, DD);
    swa_rope_q<<<(BB * SS * NH * 64) / 256, 256, 0, stream>>>(QKV, Qr);
    swa_rope_kv<<<(BB * SS * NKV * 64) / 256, 256, 0, stream>>>(QKV, Krp, Vcp);

    // attention: B*NH*(S/16) = 4096 waves
    swa_attn<<<BB * NH * (SS / 16), 32, 0, stream>>>(Qr, Krp, Vcp, Attn);

    // out = Attn @ Wo : macro-tiles = (4096/32)*(2048/64) = 4096
    swa_wmma_gemm<<<4096 / 8, 256, 0, stream>>>(Attn, Wo, nullptr, out,
                                                MM, DD, DD);
}
